// CrossModalAttention_15006615733575
// MI455X (gfx1250) — matrix operand order repaired and checked by reading, not compiled, not run on hardware
//
#include <hip/hip_runtime.h>
#include <hip/hip_bf16.h>
#include <stdint.h>

typedef __attribute__((ext_vector_type(16))) _Float16 v16h;
typedef __attribute__((ext_vector_type(8)))  _Float16 h8;
typedef __attribute__((ext_vector_type(8)))  float    v8f;
typedef __attribute__((ext_vector_type(4)))  int      v4i_t;

#define LSEQ 2048
#define NHEAD 16
#define DHEAD 64
#define HIDN 1024

#if __has_builtin(__builtin_amdgcn_global_load_async_to_lds_b128)
#define HAVE_ASYNC 1
#else
#define HAVE_ASYNC 0
#endif

typedef __attribute__((address_space(1))) v4i_t as1_v4i;
typedef __attribute__((address_space(3))) v4i_t as3_v4i;

__device__ __forceinline__ void async_ld16(const void* g, void* l) {
#if HAVE_ASYNC
    __builtin_amdgcn_global_load_async_to_lds_b128(
        (as1_v4i*)(uintptr_t)g,
        (as3_v4i*)(uint32_t)(uintptr_t)l, 0, 0);
#else
    (void)g; (void)l;
#endif
}

template <int N>
__device__ __forceinline__ void wait_async() {
#if HAVE_ASYNC
#if __has_builtin(__builtin_amdgcn_s_wait_asynccnt)
    __builtin_amdgcn_s_wait_asynccnt(N);
#else
    asm volatile("s_wait_asynccnt %0" :: "i"(N) : "memory");
#endif
#endif
}

__device__ __forceinline__ v16h frag16(h8 lo, h8 hi) {
    return __builtin_shufflevector(lo, hi, 0,1,2,3,4,5,6,7,8,9,10,11,12,13,14,15);
}

// ===========================================================================
// Pre-pass 1: f32 -> f16 elementwise (activations), 4 elems / thread
// ===========================================================================
__global__ void __launch_bounds__(256)
cvt_f16_kernel(const float* __restrict__ in, _Float16* __restrict__ out, int n) {
    const int i = (blockIdx.x * blockDim.x + threadIdx.x) * 4;
    if (i >= n) return;
    const float4 v = *(const float4*)(in + i);
    _Float16 o[4] = {(_Float16)v.x, (_Float16)v.y, (_Float16)v.z, (_Float16)v.w};
    *(__attribute__((ext_vector_type(4))) _Float16*)(out + i) =
        *(__attribute__((ext_vector_type(4))) _Float16*)o;
}

// ===========================================================================
// Pre-pass 2: W f32 [K x 1024] -> Wt f16 [1024 x K]  (LDS-tiled transpose)
// ===========================================================================
__global__ void __launch_bounds__(256)
wt_kernel(const float* __restrict__ W, _Float16* __restrict__ Wt, int K) {
    __shared__ float tile[32][33];
    const int tx = threadIdx.x & 31, ty = threadIdx.x >> 5;
    const int n0 = blockIdx.x * 32, k0 = blockIdx.y * 32;
#pragma unroll
    for (int i = 0; i < 32; i += 8)
        tile[ty + i][tx] = W[(size_t)(k0 + ty + i) * HIDN + n0 + tx];
    __syncthreads();
#pragma unroll
    for (int i = 0; i < 32; i += 8)
        Wt[(size_t)(n0 + ty + i) * K + k0 + tx] = (_Float16)tile[tx][ty + i];
}

// ===========================================================================
// GEMM: C[M,1024] = A_f16[M,K] * Wt_f16[1024,K]^T + bias
//   mode 0: f16 out, [B,H,L,64]   (q, k)
//   mode 1: f16 out, [B,H,64,L]   (v transposed for P x V)
//   mode 2: f32 out, row-major    (final projection)
// One wave computes 32(M) x 64(N): 8 accumulators, each B fragment reused
// twice, all 12 b128 loads of a k-step issued before the 8 wmmas.
// ===========================================================================
__global__ void __launch_bounds__(256)
gemm_f16_kernel(const _Float16* __restrict__ A, const _Float16* __restrict__ Wt,
                const float* __restrict__ bias, _Float16* __restrict__ out16,
                float* __restrict__ out32, int M, int K, int mode) {
    const int wave = (blockIdx.x * (blockDim.x >> 5)) + (threadIdx.x >> 5);
    const int tm = wave >> 4;           // M/32 tiles
    const int tn = wave & 15;           // 16 n-tiles of 64
    if (tm * 32 >= M) return;

    const int lane = threadIdx.x & 31;
    const int nl   = lane & 15;
    const int half = lane >> 4;

    const _Float16* A0 = A + (size_t)(tm * 32 + nl) * K;
    const _Float16* A1 = A0 + (size_t)16 * K;
    const _Float16* Wr[4];
#pragma unroll
    for (int j = 0; j < 4; ++j)
        Wr[j] = Wt + (size_t)(tn * 64 + j * 16 + nl) * K;

    v8f acc[2][4] = {};
    for (int k0 = 0; k0 < K; k0 += 32) {
        // ---- issue all fragment loads for this k-step first
        const v16h a0 = frag16(*(const h8*)(A0 + k0 + 8 * half),
                               *(const h8*)(A0 + k0 + 16 + 8 * half));
        const v16h a1 = frag16(*(const h8*)(A1 + k0 + 8 * half),
                               *(const h8*)(A1 + k0 + 16 + 8 * half));
        v16h b[4];
#pragma unroll
        for (int j = 0; j < 4; ++j)
            b[j] = frag16(*(const h8*)(Wr[j] + k0 + 8 * half),
                          *(const h8*)(Wr[j] + k0 + 16 + 8 * half));
        // ---- then 8 back-to-back wmmas
#pragma unroll
        for (int j = 0; j < 4; ++j)
            acc[0][j] = __builtin_amdgcn_wmma_f32_16x16x32_f16(
                false, a0, false, b[j], (short)0, acc[0][j], false, false);
#pragma unroll
        for (int j = 0; j < 4; ++j)
            acc[1][j] = __builtin_amdgcn_wmma_f32_16x16x32_f16(
                false, a1, false, b[j], (short)0, acc[1][j], false, false);
    }
#pragma unroll
    for (int g = 0; g < 2; ++g)
#pragma unroll
        for (int j = 0; j < 4; ++j) {
            const int ncol = tn * 64 + j * 16 + nl;
            const float bv = bias[ncol];
            const int h = ncol >> 6, d = ncol & 63;
#pragma unroll
            for (int r = 0; r < 8; ++r) {
                const int m  = tm * 32 + g * 16 + r + 8 * half;
                const float val = acc[g][j][r] + bv;
                if (mode == 0) {
                    const int bb = m >> 11, l = m & (LSEQ - 1);
                    out16[(((size_t)bb * NHEAD + h) * LSEQ + l) * DHEAD + d] = (_Float16)val;
                } else if (mode == 1) {
                    const int bb = m >> 11, l = m & (LSEQ - 1);
                    out16[(((size_t)bb * NHEAD + h) * DHEAD + d) * LSEQ + l] = (_Float16)val;
                } else {
                    out32[(size_t)m * HIDN + ncol] = val;
                }
            }
        }
}

// ===========================================================================
// Flash attention: one wave per 16-query tile, 4 waves / block.
// K/V chunks (32 keys) double-buffered in LDS via CDNA5 async-to-LDS:
// stage chunk i+1 while computing chunk i; s_wait_asynccnt 16 retires chunk i
// (async loads complete in order).
// qh: [B,H,L,64] f16, kh: [B,H,L,64] f16, vt: [B,H,64,L] f16 (transposed).
// ===========================================================================
__global__ void __launch_bounds__(128)
flash_attn_kernel(const _Float16* __restrict__ qh, const _Float16* __restrict__ kh,
                  const _Float16* __restrict__ vt, const int* __restrict__ query_mask,
                  _Float16* __restrict__ ctx) {
    __shared__ __align__(16) _Float16 sP[4][16 * 32];
#if HAVE_ASYNC
    __shared__ __align__(16) _Float16 sK[4][2][32 * 64];
    __shared__ __align__(16) _Float16 sV[4][2][64 * 32];
#endif

    const int wave = threadIdx.x >> 5;
    const int lane = threadIdx.x & 31;
    const int nl   = lane & 15;
    const int half = lane >> 4;

    const int bh = blockIdx.x;
    const int b  = bh >> 4, h = bh & 15;
    const int q0 = (blockIdx.y * 4 + wave) * 16;

    const _Float16* Q  = qh + ((size_t)bh * LSEQ + q0) * DHEAD;
    const _Float16* Kh = kh + (size_t)bh * LSEQ * DHEAD;
    const _Float16* Vt = vt + (size_t)bh * DHEAD * LSEQ;
    const int*      qm = query_mask + (size_t)b * LSEQ;

#if HAVE_ASYNC
    const int vrow = lane >> 2, vseg = lane & 3;
    auto stage = [&](int kbase, int buf) {
        // K chunk: 32x64 f16, 4KB contiguous in global
        const char* kg = (const char*)(Kh + (size_t)kbase * DHEAD);
        char*       kl = (char*)(sK[wave][buf]);
#pragma unroll
        for (int i = 0; i < 8; ++i)
            async_ld16(kg + (i * 32 + lane) * 16, kl + (i * 32 + lane) * 16);
        // V^T chunk: 64 rows x 32 keys, global row stride LSEQ f16
        const char* vg = (const char*)(Vt + kbase);
        char*       vl = (char*)(sV[wave][buf]);
#pragma unroll
        for (int i = 0; i < 8; ++i)
            async_ld16(vg + (size_t)(i * 8 + vrow) * (LSEQ * 2) + vseg * 16,
                       vl + (i * 8 + vrow) * 64 + vseg * 16);
    };
    stage(0, 0);
#endif

    // Q A-fragments for the two K=32 halves of the d=64 contraction
    const _Float16* Qr = Q + nl * DHEAD;
    v16h aq[2];
#pragma unroll
    for (int s = 0; s < 2; ++s)
        aq[s] = frag16(*(const h8*)(Qr + s * 32 + 8 * half),
                       *(const h8*)(Qr + s * 32 + 16 + 8 * half));

    float Mrow[8], Lrow[8];
#pragma unroll
    for (int r = 0; r < 8; ++r) { Mrow[r] = -1e30f; Lrow[r] = 0.f; }
    v8f O[4] = {};

    int cur = 0;
    for (int kb = 0; kb < LSEQ; kb += 32) {
        __syncthreads();   // retire previous iteration's LDS reads (sP/sK/sV)

#if HAVE_ASYNC
        const bool more = (kb + 32) < LSEQ;
        if (more) {
            stage(kb + 32, cur ^ 1);   // prefetch next chunk into alternate buffer
            wait_async<16>();          // chunk `cur` fully landed (in-order)
        } else {
            wait_async<0>();
        }
        __syncthreads();
#endif
        // ---- scores: two 16x16 key subtiles, K=64 contraction (b128 loads)
        v8f S[2] = {};
#pragma unroll
        for (int kn = 0; kn < 2; ++kn) {
#if HAVE_ASYNC
            const _Float16* Krow = sK[wave][cur] + (kn * 16 + nl) * DHEAD;
#else
            const _Float16* Krow = Kh + (size_t)(kb + kn * 16 + nl) * DHEAD;
#endif
#pragma unroll
            for (int s = 0; s < 2; ++s) {
                const v16h bkf = frag16(*(const h8*)(Krow + s * 32 + 8 * half),
                                        *(const h8*)(Krow + s * 32 + 16 + 8 * half));
                S[kn] = __builtin_amdgcn_wmma_f32_16x16x32_f16(
                    false, aq[s], false, bkf, (short)0, S[kn], false, false);
            }
        }

        // ---- mask (query_mask masks the KEY axis per reference) + scale
        const float ok0 = qm[kb + nl]      ? 1.f : 0.f;
        const float ok1 = qm[kb + 16 + nl] ? 1.f : 0.f;
        float rowmax[8];
#pragma unroll
        for (int r = 0; r < 8; ++r) {
            S[0][r] = ok0 ? S[0][r] * 0.125f : -1e30f;
            S[1][r] = ok1 ? S[1][r] * 0.125f : -1e30f;
            rowmax[r] = fmaxf(S[0][r], S[1][r]);
        }
#pragma unroll
        for (int off = 8; off > 0; off >>= 1)
#pragma unroll
            for (int r = 0; r < 8; ++r)
                rowmax[r] = fmaxf(rowmax[r], __shfl_xor(rowmax[r], off, 32));

        // ---- online softmax; P into LDS (C-layout -> row major)
        _Float16* pl = sP[wave];
        float ps[8];
#pragma unroll
        for (int r = 0; r < 8; ++r) {
            const float Mn   = fmaxf(Mrow[r], rowmax[r]);
            const float corr = __expf(Mrow[r] - Mn);
            Mrow[r] = Mn;
            Lrow[r] *= corr;
#pragma unroll
            for (int j = 0; j < 4; ++j) O[j][r] *= corr;
            const float p0 = (S[0][r] <= -1e29f) ? 0.f : __expf(S[0][r] - Mn);
            const float p1 = (S[1][r] <= -1e29f) ? 0.f : __expf(S[1][r] - Mn);
            ps[r] = p0 + p1;
            pl[(r + 8 * half) * 32 + nl]      = (_Float16)p0;
            pl[(r + 8 * half) * 32 + 16 + nl] = (_Float16)p1;
        }
#pragma unroll
        for (int off = 8; off > 0; off >>= 1)
#pragma unroll
            for (int r = 0; r < 8; ++r) ps[r] += __shfl_xor(ps[r], off, 32);
#pragma unroll
        for (int r = 0; r < 8; ++r) Lrow[r] += ps[r];

        __syncthreads();
        // ---- reload P as A-fragment (16 q x 32 k), two ds_load_b128
        const _Float16* prow = pl + nl * 32;
        const v16h ap = frag16(*(const h8*)(prow + 8 * half),
                               *(const h8*)(prow + 16 + 8 * half));

        // ---- O += P x V: load all four V B-fragments, then 4 wmmas
        v16h bvf[4];
#pragma unroll
        for (int j = 0; j < 4; ++j) {
#if HAVE_ASYNC
            const _Float16* Vrow = sV[wave][cur] + (j * 16 + nl) * 32;
#else
            const _Float16* Vrow = Vt + (size_t)(j * 16 + nl) * LSEQ + kb;
#endif
            bvf[j] = frag16(*(const h8*)(Vrow + 8 * half),
                            *(const h8*)(Vrow + 16 + 8 * half));
        }
#pragma unroll
        for (int j = 0; j < 4; ++j)
            O[j] = __builtin_amdgcn_wmma_f32_16x16x32_f16(
                false, ap, false, bvf[j], (short)0, O[j], false, false);

        cur ^= 1;
    }

    // ---- normalize rows, store ctx[b, l, h*64+d] f16 (GEMM-A layout)
    float inv[8];
#pragma unroll
    for (int r = 0; r < 8; ++r) inv[r] = Lrow[r] > 0.f ? 1.f / Lrow[r] : 0.f;
#pragma unroll
    for (int j = 0; j < 4; ++j)
#pragma unroll
        for (int r = 0; r < 8; ++r) {
            const int l = q0 + r + 8 * half;
            ctx[((size_t)b * LSEQ + l) * HIDN + h * 64 + j * 16 + nl] =
                (_Float16)(O[j][r] * inv[r]);
        }
}

// ===========================================================================
// Host launcher
// inputs: 0 query, 1 key, 2 value, 3 Wq, 4 bq, 5 Wk, 6 bk, 7 Wv, 8 bv,
//         9 Wo, 10 bo, 11 query_mask, 12 key_mask
// ===========================================================================
extern "C" void kernel_launch(void* const* d_in, const int* in_sizes, int n_in,
                              void* d_out, int out_size, void* d_ws, size_t ws_size,
                              hipStream_t stream) {
    (void)in_sizes; (void)n_in; (void)out_size; (void)ws_size;
    const float* query = (const float*)d_in[0];
    const float* key   = (const float*)d_in[1];
    const float* value = (const float*)d_in[2];
    const float* Wq = (const float*)d_in[3];
    const float* bq = (const float*)d_in[4];
    const float* Wk = (const float*)d_in[5];
    const float* bk = (const float*)d_in[6];
    const float* Wv = (const float*)d_in[7];
    const float* bv = (const float*)d_in[8];
    const float* Wo = (const float*)d_in[9];
    const float* bo = (const float*)d_in[10];
    const int* query_mask = (const int*)d_in[11];
    float* out = (float*)d_out;

    const int M = 2 * LSEQ;                                   // 4096
    const size_t headE = (size_t)2 * NHEAD * LSEQ * DHEAD;    // 4M f16

    _Float16* p   = (_Float16*)d_ws;
    _Float16* qa  = p;  p += (size_t)M * 1024;     // query f16
    _Float16* ka  = p;  p += (size_t)M * 768;      // key f16
    _Float16* va  = p;  p += (size_t)M * 768;      // value f16
    _Float16* WqT = p;  p += (size_t)1024 * 1024;
    _Float16* WkT = p;  p += (size_t)1024 * 768;
    _Float16* WvT = p;  p += (size_t)1024 * 768;
    _Float16* WoT = p;  p += (size_t)1024 * 1024;
    _Float16* qh  = p;  p += headE;
    _Float16* kh  = p;  p += headE;
    _Float16* vt  = p;  p += headE;                // [B,H,64,L]
    _Float16* ctx = p;  p += (size_t)M * 1024;

    // ---- pre-passes: f16 conversion + weight transposes
    cvt_f16_kernel<<<(M * 1024 / 4 + 255) / 256, 256, 0, stream>>>(query, qa, M * 1024);
    cvt_f16_kernel<<<(M * 768  / 4 + 255) / 256, 256, 0, stream>>>(key,   ka, M * 768);
    cvt_f16_kernel<<<(M * 768  / 4 + 255) / 256, 256, 0, stream>>>(value, va, M * 768);
    wt_kernel<<<dim3(32, 1024 / 32), 256, 0, stream>>>(Wq, WqT, 1024);
    wt_kernel<<<dim3(32,  768 / 32), 256, 0, stream>>>(Wk, WkT, 768);
    wt_kernel<<<dim3(32,  768 / 32), 256, 0, stream>>>(Wv, WvT, 768);
    wt_kernel<<<dim3(32, 1024 / 32), 256, 0, stream>>>(Wo, WoT, 1024);

    // ---- projections: (M/32)*16 = 2048 wave-tiles of 32x64 -> 256 blocks
    gemm_f16_kernel<<<256, 256, 0, stream>>>(qa, WqT, bq, qh, nullptr, M, 1024, 0);
    gemm_f16_kernel<<<256, 256, 0, stream>>>(ka, WkT, bk, kh, nullptr, M, 768, 0);
    gemm_f16_kernel<<<256, 256, 0, stream>>>(va, WvT, bv, vt, nullptr, M, 768, 1);

    // ---- attention: 32 (b,h) x 32 blocks of 4 query tiles
    flash_attn_kernel<<<dim3(32, 32), 128, 0, stream>>>(qh, kh, vt, query_mask, ctx);

    // ---- output projection -> f32 d_out
    gemm_f16_kernel<<<256, 256, 0, stream>>>(ctx, WoT, bo, nullptr, out, M, 1024, 2);
}
